// Linear_bin_79551384256886
// MI455X (gfx1250) — matrix operand order repaired and checked by reading, not compiled, not run on hardware
//
#include <hip/hip_runtime.h>
#include <hip/hip_bf16.h>

// ---------------------------------------------------------------------------
// Types for CDNA5 WMMA (wave32): v_wmma_f32_16x16x32_bf16
// ---------------------------------------------------------------------------
typedef __attribute__((ext_vector_type(16))) __bf16 v16bf;
typedef __attribute__((ext_vector_type(8)))  float  v8f;
typedef __attribute__((ext_vector_type(4)))  int    v4i;

union FragAB { v16bf v; uint4 q[2]; };
union FragC  { v8f  v; float f[8]; };

__device__ __forceinline__ unsigned short f2bf(float f) {
  unsigned int u = __builtin_bit_cast(unsigned int, f);
  u += 0x7fffu + ((u >> 16) & 1u);           // round-to-nearest-even
  return (unsigned short)(u >> 16);
}
__device__ __forceinline__ float bf2f(unsigned short h) {
  unsigned int u = ((unsigned int)h) << 16;
  return __builtin_bit_cast(float, u);
}
__device__ __forceinline__ unsigned int pk(unsigned short a, unsigned short b) {
  return (unsigned int)a | ((unsigned int)b << 16);
}

// ---------------------------------------------------------------------------
// CDNA5 async global->LDS copy (GLOBAL_LOAD_ASYNC_TO_LDS_B128, ASYNCcnt),
// with a synchronous fallback if the builtin is unavailable.
// ---------------------------------------------------------------------------
#if __has_builtin(__builtin_amdgcn_global_load_async_to_lds_b128)
#define CDNA5_ASYNC_LDS 1
#else
#define CDNA5_ASYNC_LDS 0
#endif

__device__ __forceinline__ void copy16_g2l(const void* gsrc, void* ldst) {
#if CDNA5_ASYNC_LDS
  __builtin_amdgcn_global_load_async_to_lds_b128(
      (__attribute__((address_space(1))) v4i*)gsrc,
      (__attribute__((address_space(3))) v4i*)ldst, 0, 0);
#else
  *(uint4*)ldst = *(const uint4*)gsrc;
#endif
}

__device__ __forceinline__ void wait_async_copies() {
#if CDNA5_ASYNC_LDS
#if __has_builtin(__builtin_amdgcn_s_wait_asynccnt)
  __builtin_amdgcn_s_wait_asynccnt(0);
#else
  asm volatile("s_wait_asynccnt 0" ::: "memory");
#endif
#endif
}

// ---------------------------------------------------------------------------
// Binarize weights: sign(w) in {-1,0,+1}, emitted as bf16 (exact)
// ---------------------------------------------------------------------------
__global__ void binarize_kernel(const float* __restrict__ w,
                                unsigned short* __restrict__ out, int total) {
  int i = blockIdx.x * blockDim.x + threadIdx.x;
  if (i >= total) return;
  float v = w[i];
  out[i] = v > 0.f ? (unsigned short)0x3F80u
                   : (v < 0.f ? (unsigned short)0xBF80u : (unsigned short)0u);
}

__global__ void zero_kernel(float* __restrict__ p, int n) {
  int i = blockIdx.x * blockDim.x + threadIdx.x;
  if (i < n) p[i] = 0.f;
}

// ---------------------------------------------------------------------------
// GEMM: Y[M,N] = A[M,K] * W[N,K]^T + bias[N]   (A f32 or bf16, W bf16 +-1)
// 128x128 block tile, 8 waves (2x4), each wave 64x32 = 4x2 WMMA 16x16 tiles.
// Double-buffered LDS, K-step = 32. bf16 operands staged with async
// global->LDS copies (ASYNCcnt); the f32 activation path (layer 1) converts
// f32->bf16 through VGPRs while staging.
// ---------------------------------------------------------------------------
template <bool A_F32>
__global__ __launch_bounds__(256, 2)
void gemm_bin_wmma(const void* __restrict__ Aptr,
                   const unsigned short* __restrict__ W,
                   const float* __restrict__ bias,
                   float* __restrict__ Y, int K, int N) {
  __shared__ unsigned short sA[2][128 * 32];
  __shared__ unsigned short sB[2][128 * 32];

  const int tid   = threadIdx.x;
  const int lane  = tid & 31;
  const int wave  = tid >> 5;
  const int waveM = wave >> 2;          // 0..1
  const int waveN = wave & 3;           // 0..3
  const int m0 = blockIdx.y * 128;
  const int n0 = blockIdx.x * 128;

  const int ldRow = tid >> 1;           // 0..127
  const int ldSeg = (tid & 1) << 4;     // 0 or 16 (elements)

  FragC acc[4][2];
#pragma unroll
  for (int mi = 0; mi < 4; ++mi)
#pragma unroll
    for (int ni = 0; ni < 2; ++ni)
#pragma unroll
      for (int r = 0; r < 8; ++r) acc[mi][ni].f[r] = 0.f;

  auto stage = [&](int buf, int k0) {
    // B operand (binarized weights, bf16): async DMA into LDS
    {
      const unsigned short* ws = W + (size_t)(n0 + ldRow) * K + k0 + ldSeg;
      unsigned short* ld = &sB[buf][ldRow * 32 + ldSeg];
      copy16_g2l(ws, ld);
      copy16_g2l(ws + 8, ld + 8);
    }
    if (A_F32) {
      // Layer 1: f32 -> bf16 conversion through VGPRs, then ds_store
      const float* A = (const float*)Aptr;
      const float4* src = (const float4*)(A + (size_t)(m0 + ldRow) * K + k0 + ldSeg);
      unsigned short t[16];
#pragma unroll
      for (int v = 0; v < 4; ++v) {
        float4 f = src[v];
        t[4 * v + 0] = f2bf(f.x); t[4 * v + 1] = f2bf(f.y);
        t[4 * v + 2] = f2bf(f.z); t[4 * v + 3] = f2bf(f.w);
      }
      uint4* dst = (uint4*)&sA[buf][ldRow * 32 + ldSeg];
      dst[0] = make_uint4(pk(t[0], t[1]), pk(t[2], t[3]), pk(t[4], t[5]), pk(t[6], t[7]));
      dst[1] = make_uint4(pk(t[8], t[9]), pk(t[10], t[11]), pk(t[12], t[13]), pk(t[14], t[15]));
    } else {
      // Layers 2/3: activations already bf16 -> async DMA into LDS
      const unsigned short* as = (const unsigned short*)Aptr + (size_t)(m0 + ldRow) * K + k0 + ldSeg;
      unsigned short* ld = &sA[buf][ldRow * 32 + ldSeg];
      copy16_g2l(as, ld);
      copy16_g2l(as + 8, ld + 8);
    }
  };

  stage(0, 0);
  wait_async_copies();
  __syncthreads();

  const int nk = K >> 5;
  for (int ks = 0; ks < nk; ++ks) {
    const int buf = ks & 1;
    if (ks + 1 < nk) stage(buf ^ 1, (ks + 1) << 5);

    // B fragments (32x16): lanes 0-15 -> N=lane, K{0..7,16..23}; lanes 16-31 -> K{8..15,24..31}
    FragAB bfr[2];
#pragma unroll
    for (int ni = 0; ni < 2; ++ni) {
      const int r  = waveN * 32 + ni * 16 + (lane & 15);
      const int kb = (lane & 16) ? 8 : 0;
      const uint4* p = (const uint4*)&sB[buf][r * 32 + kb];
      bfr[ni].q[0] = p[0]; bfr[ni].q[1] = p[2];
    }
#pragma unroll
    for (int mi = 0; mi < 4; ++mi) {
      // A fragment (16x32): lanes 0-15 -> M=lane, K{0..7,16..23}; lanes 16-31 -> K{8..15,24..31}
      const int r  = waveM * 64 + mi * 16 + (lane & 15);
      const int kb = (lane & 16) ? 8 : 0;
      FragAB afr;
      afr.q[0] = *(const uint4*)&sA[buf][r * 32 + kb];
      afr.q[1] = *(const uint4*)&sA[buf][r * 32 + kb + 16];
#pragma unroll
      for (int ni = 0; ni < 2; ++ni) {
        acc[mi][ni].v = __builtin_amdgcn_wmma_f32_16x16x32_bf16(
            false, afr.v, false, bfr[ni].v, (short)0, acc[mi][ni].v, false, false);
      }
    }
    // Drain this wave's async copies into LDS, then publish to the workgroup.
    wait_async_copies();
    __syncthreads();
  }

  // Epilogue: C layout -> VGPR r: lanes 0-15 give M=base+r, N=lane; lanes 16-31 M=base+8+r
#pragma unroll
  for (int ni = 0; ni < 2; ++ni) {
    const int gcol = n0 + waveN * 32 + ni * 16 + (lane & 15);
    const float bv = bias[gcol];
#pragma unroll
    for (int mi = 0; mi < 4; ++mi) {
      const int grow = m0 + waveM * 64 + mi * 16 + ((lane & 16) ? 8 : 0);
#pragma unroll
      for (int r = 0; r < 8; ++r) {
        Y[(size_t)(grow + r) * N + gcol] = acc[mi][ni].f[r] + bv;
      }
    }
  }
}

// ---------------------------------------------------------------------------
// Per-column batch stats: partial sums over 256-row chunks, global f32 atomics
// ---------------------------------------------------------------------------
__global__ __launch_bounds__(256)
void colstats_kernel(const float* __restrict__ Y, float* __restrict__ sum,
                     float* __restrict__ sumsq, int N, int rowsPerBlock) {
  const int c  = blockIdx.x * blockDim.x + threadIdx.x;
  const int r0 = blockIdx.y * rowsPerBlock;
  float s = 0.f, s2 = 0.f;
  for (int r = 0; r < rowsPerBlock; ++r) {
    float v = Y[(size_t)(r0 + r) * N + c];
    s += v; s2 += v * v;
  }
  atomicAdd(&sum[c], s);
  atomicAdd(&sumsq[c], s2);
}

// ---------------------------------------------------------------------------
// Fused BatchNorm (biased batch stats) + hardtanh + pack to bf16
// ---------------------------------------------------------------------------
__global__ __launch_bounds__(256)
void bn_htanh_kernel(const float* __restrict__ Y, const float* __restrict__ sum,
                     const float* __restrict__ sumsq, const float* __restrict__ g,
                     const float* __restrict__ be, unsigned short* __restrict__ H,
                     int N, int total, float invB) {
  int i = blockIdx.x * blockDim.x + threadIdx.x;
  if (i >= total) return;
  const int c = i % N;
  const float m   = sum[c] * invB;
  const float var = fmaxf(sumsq[c] * invB - m * m, 0.f);
  float v = (Y[i] - m) * rsqrtf(var + 1e-5f) * g[c] + be[c];
  v = fminf(fmaxf(v, -1.f), 1.f);
  H[i] = f2bf(v);
}

// ---------------------------------------------------------------------------
// Head: out[m] = sigmoid(dot(H[m, 0:512], w4) + b4), one wave32 per row
// ---------------------------------------------------------------------------
__global__ __launch_bounds__(128)
void head_kernel(const unsigned short* __restrict__ H, const float* __restrict__ w4,
                 const float* __restrict__ b4, float* __restrict__ out) {
  const int lane = threadIdx.x & 31;
  const int row  = blockIdx.x * 4 + (threadIdx.x >> 5);
  const unsigned short* h = H + (size_t)row * 512;
  float acc = 0.f;
#pragma unroll
  for (int j = 0; j < 16; ++j) {
    const int idx = lane + j * 32;
    acc += bf2f(h[idx]) * w4[idx];
  }
#pragma unroll
  for (int off = 16; off > 0; off >>= 1) acc += __shfl_down(acc, off, 32);
  if (lane == 0) out[row] = 1.f / (1.f + __expf(-(acc + b4[0])));
}

// ---------------------------------------------------------------------------
// Launcher
// ---------------------------------------------------------------------------
extern "C" void kernel_launch(void* const* d_in, const int* in_sizes, int n_in,
                              void* d_out, int out_size, void* d_ws, size_t ws_size,
                              hipStream_t stream) {
  (void)in_sizes; (void)n_in; (void)out_size; (void)ws_size;
  constexpr int Bm = 16384, D0 = 4096, H1 = 2048, H2 = 1024, H3 = 512;

  const float* x   = (const float*)d_in[0];
  const float* w1  = (const float*)d_in[1];  const float* b1  = (const float*)d_in[2];
  const float* g1  = (const float*)d_in[3];  const float* be1 = (const float*)d_in[4];
  const float* w2  = (const float*)d_in[5];  const float* b2  = (const float*)d_in[6];
  const float* g2  = (const float*)d_in[7];  const float* be2 = (const float*)d_in[8];
  const float* w3  = (const float*)d_in[9];  const float* b3  = (const float*)d_in[10];
  const float* g3  = (const float*)d_in[11]; const float* be3 = (const float*)d_in[12];
  const float* w4  = (const float*)d_in[13]; const float* b4  = (const float*)d_in[14];

  char* ws = (char*)d_ws;
  size_t off = 0;
  unsigned short* Wb1 = (unsigned short*)(ws + off); off += (size_t)H1 * D0 * 2;  // 16 MB
  unsigned short* Wb2 = (unsigned short*)(ws + off); off += (size_t)H2 * H1 * 2;  //  4 MB
  unsigned short* Wb3 = (unsigned short*)(ws + off); off += (size_t)H3 * H2 * 2;  //  1 MB
  float*          st  = (float*)(ws + off);          off += 4 * H1 * 4;           // stats (sum|sumsq)
  float*          Y   = (float*)(ws + off);          off += (size_t)Bm * H1 * 4;  // 134 MB (reused)
  unsigned short* Hb  = (unsigned short*)(ws + off);                              // 67 MB (reused)

  const float invB = 1.f / (float)Bm;

  // Binarize weights once per call
  binarize_kernel<<<(H1 * D0 + 255) / 256, 256, 0, stream>>>(w1, Wb1, H1 * D0);
  binarize_kernel<<<(H2 * H1 + 255) / 256, 256, 0, stream>>>(w2, Wb2, H2 * H1);
  binarize_kernel<<<(H3 * H2 + 255) / 256, 256, 0, stream>>>(w3, Wb3, H3 * H2);

  // ---- Layer 1: x(f32) @ sign(w1)^T + b1 -> BN -> htanh -> Hb(bf16)
  gemm_bin_wmma<true><<<dim3(H1 / 128, Bm / 128), 256, 0, stream>>>(x, Wb1, b1, Y, D0, H1);
  zero_kernel<<<(2 * H1 + 255) / 256, 256, 0, stream>>>(st, 2 * H1);
  colstats_kernel<<<dim3(H1 / 256, Bm / 256), 256, 0, stream>>>(Y, st, st + H1, H1, 256);
  bn_htanh_kernel<<<(Bm * H1 + 255) / 256, 256, 0, stream>>>(Y, st, st + H1, g1, be1, Hb, H1, Bm * H1, invB);

  // ---- Layer 2
  gemm_bin_wmma<false><<<dim3(H2 / 128, Bm / 128), 256, 0, stream>>>(Hb, Wb2, b2, Y, H1, H2);
  zero_kernel<<<(2 * H2 + 255) / 256, 256, 0, stream>>>(st, 2 * H2);
  colstats_kernel<<<dim3(H2 / 256, Bm / 256), 256, 0, stream>>>(Y, st, st + H2, H2, 256);
  bn_htanh_kernel<<<(Bm * H2 + 255) / 256, 256, 0, stream>>>(Y, st, st + H2, g2, be2, Hb, H2, Bm * H2, invB);

  // ---- Layer 3 (dropout is eval-mode identity)
  gemm_bin_wmma<false><<<dim3(H3 / 128, Bm / 128), 256, 0, stream>>>(Hb, Wb3, b3, Y, H2, H3);
  zero_kernel<<<(2 * H3 + 255) / 256, 256, 0, stream>>>(st, 2 * H3);
  colstats_kernel<<<dim3(H3 / 256, Bm / 256), 256, 0, stream>>>(Y, st, st + H3, H3, 256);
  bn_htanh_kernel<<<(Bm * H3 + 255) / 256, 256, 0, stream>>>(Y, st, st + H3, g3, be3, Hb, H3, Bm * H3, invB);

  // ---- Head: 512 -> 1, sigmoid
  head_kernel<<<Bm / 4, 128, 0, stream>>>(Hb, w4, b4, (float*)d_out);
}
